// RelationGAT_57887569215725
// MI455X (gfx1250) — matrix operand order repaired and checked by reading, not compiled, not run on hardware
//
#include <hip/hip_runtime.h>
#include <hip/hip_bf16.h>

// ---------------------------------------------------------------------------
// RelationGAT fused attention for MI455X (gfx1250, wave32, WMMA), fully
// transposed dataflow so every WMMA C-result feeds the next WMMA's B operand
// through register cross-lane ops only (no LDS, no barriers):
//   qT = Wq @ xT + bq ; ST = keys @ qT ; PT = online-softmax(ST)
//   OT += VT @ PT ; outT = Wv @ (OT/l) + bv
// ---------------------------------------------------------------------------

typedef __attribute__((ext_vector_type(16))) __bf16        v16bf;
typedef __attribute__((ext_vector_type(8)))  float         v8f;
typedef __attribute__((ext_vector_type(4)))  unsigned int  u32x4;
typedef __attribute__((ext_vector_type(4)))  float         f32x4;

__device__ __forceinline__ unsigned short f2bf(float f) {
  unsigned int x = __float_as_uint(f);
  unsigned int r = x + 0x7FFFu + ((x >> 16) & 1u);   // round-to-nearest-even
  return (unsigned short)(r >> 16);
}

// Pack two f32 -> bf16x2 in one instruction: v_cvt_pk_bf16_f32 (RNE) if the
// toolchain has it, else a single v_perm_b32 (truncating bf16).
__device__ __forceinline__ unsigned int pkbf2(float a, float b) {
#if __has_builtin(__builtin_amdgcn_cvt_pk_bf16_f32)
  typedef __attribute__((ext_vector_type(2))) __bf16 v2bf;
  union { v2bf v; unsigned int u; } u;
  u.v = __builtin_amdgcn_cvt_pk_bf16_f32(a, b);
  return u.u;
#elif __has_builtin(__builtin_amdgcn_perm)
  // result bytes: [1:0] = a[3:2] (lo bf16), [3:2] = b[3:2] (hi bf16)
  return __builtin_amdgcn_perm(__float_as_uint(b), __float_as_uint(a),
                               0x07060302u);
#else
  return (unsigned int)f2bf(a) | ((unsigned int)f2bf(b) << 16);
#endif
}

// Exchange a dword with lane^16 (the other 16-lane half) as pure VALU.
__device__ __forceinline__ unsigned int xor16u(unsigned int v) {
#if __has_builtin(__builtin_amdgcn_permlanex16)
  return __builtin_amdgcn_permlanex16(v, v, 0x76543210u, 0xfedcba98u,
                                      false, false);
#else
  return (unsigned int)__shfl_xor((int)v, 16, 32);
#endif
}
__device__ __forceinline__ float xor16f(float v) {
  return __uint_as_float(xor16u(__float_as_uint(v)));
}

// A fragment (16x32 bf16): lane half h reads K = [kOff..kOff+16) of its
// row contiguously (kOff = 16*h); caller pre-offsets p by kOff.
__device__ __forceinline__ v16bf frag_a16(const unsigned short* p) {
  union { u32x4 q[2]; v16bf v; } u;
  u.q[0] = *(const u32x4*)(p);
  u.q[1] = *(const u32x4*)(p + 8);
  return u.v;
}

// B fragment (32x16 bf16) from 16 *contiguous* f32 (lane half h pre-offset
// by 16*h): B lane holds its column's K=[16h..16h+16) sequentially.
__device__ __forceinline__ v16bf frag_b_f32(const float* p) {
  f32x4 a = *(const f32x4*)(p);
  f32x4 b = *(const f32x4*)(p + 4);
  f32x4 c = *(const f32x4*)(p + 8);
  f32x4 d = *(const f32x4*)(p + 12);
  union { unsigned int u[8]; v16bf v; } r;
  r.u[0] = pkbf2(a[0], a[1]); r.u[1] = pkbf2(a[2], a[3]);
  r.u[2] = pkbf2(b[0], b[1]); r.u[3] = pkbf2(b[2], b[3]);
  r.u[4] = pkbf2(c[0], c[1]); r.u[5] = pkbf2(c[2], c[3]);
  r.u[6] = pkbf2(d[0], d[1]); r.u[7] = pkbf2(d[2], d[3]);
  return r.v;
}

// Convert a pair of C-layout f32 tiles (clo = K 32s..32s+15, chi =
// K 32s+16..32s+31 of one 32-K block) into the B operand fragment.
// Pack to bf16x2 dwords first, then one permlanex16 per dword pair:
// a lane (column n, half h) owns half its column; lane n^16 owns the rest.
__device__ __forceinline__ v16bf c_pair_to_b(v8f clo, v8f chi, int h) {
  unsigned int dlo[4], dhi[4];
#pragma unroll
  for (int i = 0; i < 4; ++i) {
    dlo[i] = pkbf2(clo[2 * i], clo[2 * i + 1]);
    dhi[i] = pkbf2(chi[2 * i], chi[2 * i + 1]);
  }
  union { unsigned int u[8]; v16bf v; } r;
#pragma unroll
  for (int i = 0; i < 4; ++i) {
    unsigned int send = h ? dlo[i] : dhi[i];   // what the partner lane needs
    unsigned int recv = xor16u(send);
    r.u[i]     = h ? recv   : dlo[i];
    r.u[4 + i] = h ? dhi[i] : recv;
  }
  return r.v;
}

__device__ __forceinline__ v8f wmma_bf16(v16bf a, v16bf b, v8f c) {
  return __builtin_amdgcn_wmma_f32_16x16x32_bf16(
      false, a, false, b, (short)0, c, false, false);
}

// ---------------------------------------------------------------------------
// Prep kernel: keys = nei@Wk^T + bk (bf16, [1024][64]); Vt = nei^T (bf16,
// [64][1024]); bf16 copies of Wq, Wv. All L2-resident, reused by every block.
// ---------------------------------------------------------------------------
__global__ __launch_bounds__(256) void gat_prep(
    const float* __restrict__ nei, const float* __restrict__ wk,
    const float* __restrict__ bk, const float* __restrict__ wqf,
    const float* __restrict__ wvf, unsigned short* __restrict__ keys,
    unsigned short* __restrict__ vt, unsigned short* __restrict__ wqb,
    unsigned short* __restrict__ wvb) {
  int gid = blockIdx.x * blockDim.x + threadIdx.x;
  if (gid < 1024 * 64) {
    int m = gid >> 6, o = gid & 63;
    const float* nr = nei + m * 64;
    const float* wr = wk + o * 64;
    float s = bk[o];
#pragma unroll 8
    for (int i = 0; i < 64; ++i) s += nr[i] * wr[i];
    keys[m * 64 + o] = f2bf(s);
  } else if (gid < 2 * 1024 * 64) {
    int g = gid - 1024 * 64;
    int o = g >> 10, m = g & 1023;
    vt[o * 1024 + m] = f2bf(nei[m * 64 + o]);
  } else if (gid < 2 * 1024 * 64 + 4096) {
    int g = gid - 2 * 1024 * 64;
    wqb[g] = f2bf(wqf[g]);
  } else if (gid < 2 * 1024 * 64 + 8192) {
    int g = gid - (2 * 1024 * 64 + 4096);
    wvb[g] = f2bf(wvf[g]);
  }
}

// ---------------------------------------------------------------------------
// Main fused kernel: 8 waves/block, each wave owns 16 query columns.
// No LDS, no barriers: waves are fully independent.
// ---------------------------------------------------------------------------
__global__ __launch_bounds__(256) void gat_main(
    const float* __restrict__ x, const float* __restrict__ bq,
    const float* __restrict__ bv, const unsigned short* __restrict__ keys,
    const unsigned short* __restrict__ vt, const unsigned short* __restrict__ wq,
    const unsigned short* __restrict__ wv, float* __restrict__ out, int n) {
  const int lane = threadIdx.x & 31;
  const int wave = threadIdx.x >> 5;
  const int half = lane >> 4;      // which K/feature half of the column we own
  const int l16  = lane & 15;      // our query column within the tile
  const int kOff = half * 16;      // A-fragment contiguous K sub-block
  const int bOff = half * 16;      // B-fragment contiguous K sub-block

  const int rowBase = blockIdx.x * 128 + wave * 16;
  int qrow = rowBase + l16;
  if (qrow >= n) qrow = n - 1;     // clamp loads; keeps EXEC all-ones for WMMA

  // ---- xT as B fragments: lane's column = its query row of x (f32->bf16) --
  const float* xr = x + qrow * 64;
  v16bf xB0 = frag_b_f32(xr + 0  + bOff);
  v16bf xB1 = frag_b_f32(xr + 32 + bOff);

  // ---- qT = Wq @ xT + bq : C tiles have rows = out-feature o ----
  v8f qacc[4];
#pragma unroll
  for (int t = 0; t < 4; ++t) {
    const unsigned short* wrow = wq + (16 * t + l16) * 64;
    v16bf a0 = frag_a16(wrow + 0  + kOff);
    v16bf a1 = frag_a16(wrow + 32 + kOff);
    v8f c = {0.f, 0.f, 0.f, 0.f, 0.f, 0.f, 0.f, 0.f};
    c = wmma_bf16(a0, xB0, c);
    c = wmma_bf16(a1, xB1, c);
    f32x4 b0 = *(const f32x4*)(bq + 16 * t + 8 * half);
    f32x4 b1 = *(const f32x4*)(bq + 16 * t + 8 * half + 4);
#pragma unroll
    for (int r = 0; r < 4; ++r) { c[r] += b0[r]; c[4 + r] += b1[r]; }
    qacc[t] = c;
  }
  // qT (C layout) -> B operand fragments via lane-pair exchange
  v16bf qB0 = c_pair_to_b(qacc[0], qacc[1], half);
  v16bf qB1 = c_pair_to_b(qacc[2], qacc[3], half);

  // ---- online softmax state: one query column per lane ----
  float mrun = -3.0e38f, lrun = 0.f;
  v8f O[4];
#pragma unroll
  for (int t = 0; t < 4; ++t) O[t] = (v8f){0.f,0.f,0.f,0.f,0.f,0.f,0.f,0.f};

  // ---- stream over 1024 keys in 16 chunks of 64 ----
  for (int c = 0; c < 16; ++c) {
    // ST[nt] = keys-tile @ qT : C rows = neighbors, cols = queries (8 WMMAs)
    v8f St[4];
#pragma unroll
    for (int nt = 0; nt < 4; ++nt) {
      const unsigned short* krow = keys + (64 * c + 16 * nt + l16) * 64;
      v16bf a0 = frag_a16(krow + 0  + kOff);
      v16bf a1 = frag_a16(krow + 32 + kOff);
      v8f s = {0.f, 0.f, 0.f, 0.f, 0.f, 0.f, 0.f, 0.f};
      s = wmma_bf16(a0, qB0, s);
      s = wmma_bf16(a1, qB1, s);
      St[nt] = s;
    }

    // column-wise online softmax (lane pair n, n+16 shares a column)
    float mx = St[0][0];
#pragma unroll
    for (int nt = 0; nt < 4; ++nt)
#pragma unroll
      for (int r = 0; r < 8; ++r) mx = fmaxf(mx, St[nt][r]);
    mx = fmaxf(mx, xor16f(mx));
    float mn = fmaxf(mrun, mx);
    float sc = __expf(mrun - mn);
    float ps = 0.f;
#pragma unroll
    for (int nt = 0; nt < 4; ++nt)
#pragma unroll
      for (int r = 0; r < 8; ++r) {
        float p = __expf(St[nt][r] - mn);
        St[nt][r] = p;
        ps += p;
      }
    ps += xor16f(ps);
    lrun = lrun * sc + ps;
    mrun = mn;
#pragma unroll
    for (int t = 0; t < 4; ++t)
#pragma unroll
      for (int r = 0; r < 8; ++r) O[t][r] *= sc;

    // PT (C layout) -> B fragments, then OT += VT @ PT (8 WMMAs)
    v16bf pB0 = c_pair_to_b(St[0], St[1], half);
    v16bf pB1 = c_pair_to_b(St[2], St[3], half);
#pragma unroll
    for (int t = 0; t < 4; ++t) {
      const unsigned short* vrow = vt + (16 * t + l16) * 1024 + 64 * c;
      v16bf a0 = frag_a16(vrow + 0  + kOff);
      v16bf a1 = frag_a16(vrow + 32 + kOff);
      O[t] = wmma_bf16(a0, pB0, O[t]);
      O[t] = wmma_bf16(a1, pB1, O[t]);
    }
  }

  // ---- aggT = OT / l ; outT = Wv @ aggT + bv ----
  float inv = 1.0f / lrun;
#pragma unroll
  for (int t = 0; t < 4; ++t)
#pragma unroll
    for (int r = 0; r < 8; ++r) O[t][r] *= inv;
  v16bf gB0 = c_pair_to_b(O[0], O[1], half);
  v16bf gB1 = c_pair_to_b(O[2], O[3], half);

  const int orow  = rowBase + l16;
  const bool ok   = orow < n;
  float* op = out + orow * 64;
#pragma unroll
  for (int t = 0; t < 4; ++t) {
    const unsigned short* wrow = wv + (16 * t + l16) * 64;
    v16bf a0 = frag_a16(wrow + 0  + kOff);
    v16bf a1 = frag_a16(wrow + 32 + kOff);
    v8f c = {0.f, 0.f, 0.f, 0.f, 0.f, 0.f, 0.f, 0.f};
    c = wmma_bf16(a0, gB0, c);
    c = wmma_bf16(a1, gB1, c);
    // C element r -> out feature o = 16t + 8*half + r : 8 consecutive floats
    f32x4 b0 = *(const f32x4*)(bv + 16 * t + 8 * half);
    f32x4 b1 = *(const f32x4*)(bv + 16 * t + 8 * half + 4);
    f32x4 s0, s1;
#pragma unroll
    for (int r = 0; r < 4; ++r) { s0[r] = c[r] + b0[r]; s1[r] = c[4 + r] + b1[r]; }
    if (ok) {
      *(f32x4*)(op + 16 * t + 8 * half)     = s0;
      *(f32x4*)(op + 16 * t + 8 * half + 4) = s1;
    }
  }
}

// ---------------------------------------------------------------------------
extern "C" void kernel_launch(void* const* d_in, const int* in_sizes, int n_in,
                              void* d_out, int out_size, void* d_ws,
                              size_t ws_size, hipStream_t stream) {
  (void)n_in; (void)out_size; (void)ws_size;
  const float* x   = (const float*)d_in[0];
  const float* nei = (const float*)d_in[1];
  const float* Wq  = (const float*)d_in[2];
  const float* bq  = (const float*)d_in[3];
  const float* Wk  = (const float*)d_in[4];
  const float* bk  = (const float*)d_in[5];
  const float* Wv  = (const float*)d_in[6];
  const float* bv  = (const float*)d_in[7];
  float* out = (float*)d_out;
  const int n = in_sizes[0] / 64;

  // workspace (bf16): keys[1024*64] | vt[64*1024] | wq[64*64] | wv[64*64]
  unsigned short* keys = (unsigned short*)d_ws;
  unsigned short* vt   = keys + 1024 * 64;
  unsigned short* wqb  = vt + 64 * 1024;
  unsigned short* wvb  = wqb + 64 * 64;

  const int prep_items = 2 * 1024 * 64 + 2 * 4096;
  gat_prep<<<(prep_items + 255) / 256, 256, 0, stream>>>(
      nei, Wk, bk, Wq, Wv, keys, vt, wqb, wvb);

  const int blocks = (n + 127) / 128;
  gat_main<<<blocks, 256, 0, stream>>>(x, bq, bv, keys, vt, wqb, wvb, out, n);
}
